// GATLayerTemporal_24180665877123
// MI455X (gfx1250) — matrix operand order repaired and checked by reading, not compiled, not run on hardware
//
#include <hip/hip_runtime.h>
#include <math.h>

// ---------------------------------------------------------------------------
// GAT temporal layer, algebraically collapsed:
//   Wh2[n,t,c,v] = sum_ti W[ti,t] * h[n,c,ti,v]     (WMMA, v = N-dim, coalesced)
//   s[n,v,h] = dot(a, Wh[n,v, h-block])             (block reduction, lane = v)
//   P = exp(lrelu(s)); E = sum_h P; G = 1/(E_i+E_j)
//   per (n,t):  Y = X @ G  (X[c,j] = Wh2[n,t,c,j], *P_j for odd t)
//               even t: Y[:,i] *= P[n,i,t/2]
//               Z = Y @ adj_norm ; out = ELU(Z)      (4 WMMAs per wave)
// f32 storage (BW-bound), f16 WMMA operands, f32 accumulation.
// ---------------------------------------------------------------------------

typedef __attribute__((ext_vector_type(16))) _Float16 v16h;
typedef __attribute__((ext_vector_type(8)))  float    v8f;

#define NB 4
#define HH 16
#define TT 32
#define VV 32
#define HW 256
#define ALPHA 0.2f

// A-operand (16-bit, 16x32) K index for element e, lane-group g (contiguous k = 16g + e, matches B)
__device__ __forceinline__ int a_kidx(int e, int g) {
  return e + 16 * g;
}

__device__ __forceinline__ v8f wmma16(v16h a, v16h b, v8f c) {
  return __builtin_amdgcn_wmma_f32_16x16x32_f16(
      /*neg_a=*/false, a, /*neg_b=*/false, b,
      /*c_mod=*/(short)0, c, /*reuse_a=*/false, /*reuse_b=*/false);
}

// --------------------------- adj_norm (tiny) -------------------------------
__global__ void k_adj(const float* __restrict__ B, float* __restrict__ adjn) {
  __shared__ float A[VV * VV];
  __shared__ float R[VV * VV];
  __shared__ float rs[VV];
  int tid = threadIdx.x;                 // 1024 threads
  int i = tid >> 5, j = tid & 31;
  float x = B[tid] + 1e-6f + (i == j ? 1.f : 0.f);
  A[tid] = x;
  R[tid] = x;
  __syncthreads();
  for (int s = 512; s > 0; s >>= 1) {
    if (tid < s) R[tid] = fminf(R[tid], R[tid + s]);
    __syncthreads();
  }
  float mn = R[0];
  __syncthreads();
  R[tid] = x;
  __syncthreads();
  for (int s = 512; s > 0; s >>= 1) {
    if (tid < s) R[tid] = fmaxf(R[tid], R[tid + s]);
    __syncthreads();
  }
  float mx = R[0];
  __syncthreads();
  float an = (x - mn) / (mx - mn);
  A[tid] = an;
  __syncthreads();
  if (tid < VV) {
    float sum = 0.f;
    for (int jj = 0; jj < VV; ++jj) sum += A[tid * VV + jj];
    rs[tid] = rsqrtf(sum);
  }
  __syncthreads();
  adjn[tid] = an * rs[i] * rs[j];
}

// --------------------------- Wh2 = W^T x h (WMMA) --------------------------
// Wh2[((n*T + to)*HW + c)*V + v] = sum_ti W[ti,to] * h[n,c,ti,v]
// One wave per (n,c), computing the full 32x32 (to x v) block with 4 WMMAs.
// B-operand (h, v fastest) and D-stores are coalesced 64B half-wave rows.
__global__ void k_wh(const float* __restrict__ h, const float* __restrict__ W,
                     float* __restrict__ Wh2) {
  __shared__ _Float16 Wth[TT * TT];
  int tid = threadIdx.x;
  int wi = tid >> 5, lane = tid & 31;
  int g = lane >> 4, mr = lane & 15;
  for (int idx = tid; idx < TT * TT; idx += 256)
    Wth[idx] = (_Float16)W[idx];
  __syncthreads();

  int w = blockIdx.x * 8 + wi;           // N*HW = 1024 waves
  int n = w >> 8, c = w & 255;

  // h flat: ((n*HW + c)*T + ti)*V + v
  const float* hb = h + ((size_t)n * HW + c) * TT * VV;
  v16h alo, ahi, blo, bhi;
#pragma unroll
  for (int e = 0; e < 16; ++e) {
    int ti = a_kidx(e, g);
    alo[e] = Wth[ti * TT + mr];                         // A[M=to][K=ti], to 0-15
    ahi[e] = Wth[ti * TT + 16 + mr];                    // to 16-31
    blo[e] = (_Float16)hb[(16 * g + e) * VV + mr];      // B[K=ti][N=v], v 0-15
    bhi[e] = (_Float16)hb[(16 * g + e) * VV + 16 + mr]; // v 16-31
  }
  v8f d00 = {}, d01 = {}, d10 = {}, d11 = {};
  d00 = wmma16(alo, blo, d00);
  d01 = wmma16(alo, bhi, d01);
  d10 = wmma16(ahi, blo, d10);
  d11 = wmma16(ahi, bhi, d11);

  float* ob = Wh2 + ((size_t)n * TT * HW + c) * VV;
#pragma unroll
  for (int r = 0; r < 8; ++r) {
    int to = r + 8 * g;                                 // D[M=to][N=v]
    ob[(size_t)to * HW * VV + mr]             = d00[r];
    ob[(size_t)to * HW * VV + 16 + mr]        = d01[r];
    ob[(size_t)(to + 16) * HW * VV + mr]      = d10[r];
    ob[(size_t)(to + 16) * HW * VV + 16 + mr] = d11[r];
  }
}

// --------------------------- s -> P ----------------------------------------
// One block per (n,h2); lane = v so reads of Wh2[n,t,c,v] are coalesced.
__global__ void k_sp(const float* __restrict__ Wh2, const float* __restrict__ av,
                     float* __restrict__ P) {
  __shared__ float partial[8][VV];
  int tid = threadIdx.x;
  int wi = tid >> 5, lane = tid & 31;
  int n = blockIdx.x >> 4, h2 = blockIdx.x & 15;       // N*HH = 64 blocks
  const float* base = Wh2 + (size_t)n * TT * HW * VV +
                      (size_t)h2 * 16 * VV + lane;
  float acc = 0.f;
#pragma unroll 4
  for (int q = 0; q < 64; ++q) {
    int p = wi * 64 + q;                               // p = cl*32 + t
    int cl = p >> 5, t = p & 31;
    acc += av[p] * base[(size_t)t * HW * VV + (size_t)cl * VV];
  }
  partial[wi][lane] = acc;
  __syncthreads();
  if (tid < VV) {
    float s = 0.f;
#pragma unroll
    for (int w8 = 0; w8 < 8; ++w8) s += partial[w8][tid];
    float e = s > 0.f ? s : ALPHA * s;                 // LeakyReLU
    P[(n * VV + tid) * HH + h2] = expf(e);
  }
}

// --------------------------- E, G ------------------------------------------
__global__ void k_g(const float* __restrict__ P, float* __restrict__ G) {
  __shared__ float Es[NB * VV];
  int tid = threadIdx.x;                 // 1024 threads
  if (tid < NB * VV) {
    float e = 0.f;
    for (int h2 = 0; h2 < HH; ++h2) e += P[tid * HH + h2];
    Es[tid] = e;
  }
  __syncthreads();
  for (int idx = tid; idx < NB * VV * VV; idx += 1024) {
    int n = idx >> 10, i = (idx >> 5) & 31, j = idx & 31;
    G[idx] = 1.f / (Es[n * VV + i] + Es[n * VV + j]);
  }
}

// --------------------------- main: Y=X@G ; Z=Y@adjn ; ELU ------------------
// blockIdx/2 selects (n,t); 8 waves/block each own a 16-row M tile of c.
__global__ void k_main(const float* __restrict__ Wh2, const float* __restrict__ P,
                       const float* __restrict__ G, const float* __restrict__ adjn,
                       float* __restrict__ out) {
  __shared__ _Float16 Gh[VV * VV];
  __shared__ _Float16 Ah[VV * VV];
  __shared__ float    Pl[VV];
  __shared__ _Float16 Ysh[8][16 * 32];
  int tid = threadIdx.x;
  int wi = tid >> 5, lane = tid & 31;
  int g = lane >> 4, mr = lane & 15;
  int nt = blockIdx.x >> 1;              // uniform per block: 0..127
  int n = nt >> 5, t = nt & 31;
  int h2 = t >> 1;
  bool odd = (t & 1) != 0;
  int mt = (blockIdx.x & 1) * 8 + wi;    // 0..15
  int c0 = mt * 16;

  // This wave's contiguous 128B A row; prefetch before the staging barrier.
  const float* arow = Wh2 + ((size_t)(n * TT + t) * HW + c0 + mr) * VV;
  __builtin_prefetch(arow, 0, 1);        // global_prefetch_b8

  for (int idx = tid; idx < VV * VV; idx += 256) {
    Gh[idx] = (_Float16)G[n * VV * VV + idx];
    Ah[idx] = (_Float16)adjn[idx];
  }
  if (tid < VV) Pl[tid] = P[(n * VV + tid) * HH + h2];
  __syncthreads();

  // First GEMM: Y[c,i] = sum_j X[c,j] * G[i,j]  (G symmetric: B[K=j][N=i])
  v16h a, b0, b1;
#pragma unroll
  for (int e = 0; e < 16; ++e) {
    int j = a_kidx(e, g);
    float x = arow[j];                   // coalesced (one 64B run per lane)
    if (odd) x *= Pl[j];
    a[e]  = (_Float16)x;
    b0[e] = Gh[(16 * g + e) * VV + mr];
    b1[e] = Gh[(16 * g + e) * VV + 16 + mr];
  }
  v8f y0 = {}, y1 = {};
  y0 = wmma16(a, b0, y0);
  y1 = wmma16(a, b1, y1);

  if (!odd) {                            // even t: scale columns by P[n,i,h2]
    float p0 = Pl[mr];
    float p1 = Pl[16 + mr];
#pragma unroll
    for (int r = 0; r < 8; ++r) { y0[r] *= p0; y1[r] *= p1; }
  }

  // Stage Y (f16) through LDS: D-format -> A-format for the 2nd GEMM
#pragma unroll
  for (int r = 0; r < 8; ++r) {
    Ysh[wi][(r + 8 * g) * 32 + mr]      = (_Float16)y0[r];
    Ysh[wi][(r + 8 * g) * 32 + 16 + mr] = (_Float16)y1[r];
  }
  __syncthreads();

  v16h a2, d0, d1;
#pragma unroll
  for (int e = 0; e < 16; ++e) {
    int k = a_kidx(e, g);                // i index
    a2[e] = Ysh[wi][mr * 32 + k];
    d0[e] = Ah[(16 * g + e) * VV + mr];
    d1[e] = Ah[(16 * g + e) * VV + 16 + mr];
  }
  v8f z0 = {}, z1 = {};
  z0 = wmma16(a2, d0, z0);
  z1 = wmma16(a2, d1, z1);

  // ELU + store: out[((n*HW + c)*T + t)*V + v]  (64B half-wave rows)
  float* ob = out + (size_t)((n * HW + c0) * TT + t) * VV;
#pragma unroll
  for (int r = 0; r < 8; ++r) {
    int c = r + 8 * g;
    float v0 = z0[r], v1 = z1[r];
    v0 = v0 > 0.f ? v0 : expf(v0) - 1.f;
    v1 = v1 > 0.f ? v1 : expf(v1) - 1.f;
    ob[(size_t)c * TT * VV + mr]      = v0;
    ob[(size_t)c * TT * VV + 16 + mr] = v1;
  }
}

// ---------------------------------------------------------------------------
extern "C" void kernel_launch(void* const* d_in, const int* in_sizes, int n_in,
                              void* d_out, int out_size, void* d_ws, size_t ws_size,
                              hipStream_t stream) {
  const float* h    = (const float*)d_in[0];
  const float* W    = (const float*)d_in[1];
  const float* av   = (const float*)d_in[2];
  const float* Badj = (const float*)d_in[3];
  float* out = (float*)d_out;

  char* ws = (char*)d_ws;
  float* Wh2  = (float*)ws;                                     // 4 MiB
  float* P    = (float*)(ws + (size_t)NB * TT * HW * VV * sizeof(float));
  float* G    = P + NB * VV * HH;                               // 4096 floats
  float* adjn = G + NB * VV * VV;                               // 1024 floats

  k_adj <<<1,   1024, 0, stream>>>(Badj, adjn);
  k_wh  <<<128, 256,  0, stream>>>(h, W, Wh2);
  k_sp  <<<64,  256,  0, stream>>>(Wh2, av, P);
  k_g   <<<1,   1024, 0, stream>>>(P, G);
  k_main<<<256, 256,  0, stream>>>(Wh2, P, G, adjn, out);
}
